// MaskedAutoencoderPPG_90082644066463
// MI455X (gfx1250) — matrix operand order repaired and checked by reading, not compiled, not run on hardware
//
#include <hip/hip_runtime.h>
#include <hip/hip_bf16.h>

// ---------------- CDNA5 WMMA types ----------------
typedef __attribute__((ext_vector_type(16))) __bf16 v16bf;
typedef __attribute__((ext_vector_type(8)))  float  v8f;

// 16-bit A-matrix 16x32 (MxK): lane h = lane>>4, element e -> K
__device__ __forceinline__ int kmapA(int e, int h) {
    return e + h * 8 + ((e >= 8) ? 8 : 0);
}
// 16-bit B-matrix 32x16 (KxN): element e -> K
__device__ __forceinline__ int kmapB(int e, int h) {
    return e + h * 8 + ((e >= 8) ? 8 : 0);
}
__device__ __forceinline__ __bf16 f2bf(float f) { return (__bf16)f; }

#define EPS 1e-5f

// ---------------- 0) f32 -> bf16 weight prep ----------------
__global__ __launch_bounds__(256)
void f32_to_bf16(const float* __restrict__ in, __bf16* __restrict__ out, int n)
{
    const int i = blockIdx.x * 256 + threadIdx.x;
    if (i < n) out[i] = (__bf16)in[i];
}

// ---------------- 1) Masking (argsort via stable rank count) ----------------
__global__ __launch_bounds__(256)
void mask_kernel(const float* __restrict__ x, const float* __restrict__ noise,
                 float* __restrict__ masked_x, float* __restrict__ mask,
                 int L, int NP, int keep)
{
    __shared__ float sn[512];
    const int b = blockIdx.x;
    const float* nrow = noise + (size_t)b * NP;
    for (int i = threadIdx.x; i < NP; i += 256) sn[i] = nrow[i];
    __syncthreads();
    for (int p = threadIdx.x; p < NP; p += 256) {
        const float v = sn[p];
        int rank = 0;
        for (int j = 0; j < NP; ++j) {
            const float u = sn[j];
            rank += (u < v) || (u == v && j < p);   // stable tie-break == argsort
        }
        const float mv    = (rank < keep) ? 0.f : 1.f;
        const float keepf = 1.f - mv;
        const float* xr = x        + (size_t)b * L + p * 16;
        float*       mo = mask     + (size_t)b * L + p * 16;
        float*       xo = masked_x + (size_t)b * L + p * 16;
        #pragma unroll
        for (int t = 0; t < 16; ++t) { mo[t] = mv; xo[t] = xr[t] * keepf; }
    }
}

// ---------------- 2) conv1: input-stationary, all 16 Cout per thread, bf16 out ----
__global__ __launch_bounds__(256)
void conv1_bnrelu(const float* __restrict__ in, const float* __restrict__ w,
                  const float* __restrict__ g, const float* __restrict__ bb,
                  const float* __restrict__ mm, const float* __restrict__ vv,
                  __bf16* __restrict__ out, int L)
{
    const int l = blockIdx.x * 256 + threadIdx.x;
    const int b = blockIdx.y;
    const float* xr = in + (size_t)b * L;
    float xv[7];
    #pragma unroll
    for (int t = 0; t < 7; ++t) {
        const int li = l + t - 3;
        xv[t] = (li >= 0 && li < L) ? xr[li] : 0.f;
    }
    #pragma unroll
    for (int co = 0; co < 16; ++co) {       // weights/BN uniform -> scalar loads
        float acc = 0.f;
        #pragma unroll
        for (int t = 0; t < 7; ++t) acc += w[co * 7 + t] * xv[t];
        const float s = g[co] * __frsqrt_rn(vv[co] + EPS);
        float val = (acc - mm[co]) * s + bb[co];
        out[((size_t)b * 16 + co) * L + l] = (__bf16)fmaxf(val, 0.f);
    }
}

// ---------------- 3) Implicit-GEMM conv (k=3, pad=1), bf16 WMMA, 16x64 tile/wave ---
// Block = 4 waves; each wave owns a 16(Cout) x 64(Lout) tile (4 accumulators, A reuse).
__global__ __launch_bounds__(128)
void conv_wmma_bnrelu(const __bf16* __restrict__ in, const __bf16* __restrict__ wbf,
                      const float* __restrict__ g, const float* __restrict__ bb,
                      const float* __restrict__ mm, const float* __restrict__ vv,
                      __bf16* __restrict__ out,
                      int Cin, int Cout, int Lin, int Lout, int stride)
{
    const int KT   = Cin * 3;
    const int wave = threadIdx.x >> 5;
    const int lane = threadIdx.x & 31;
    const int lm   = lane & 15;
    const int h    = lane >> 4;
    const int ctile = blockIdx.y;
    const int b     = blockIdx.z;
    const int lbase = (blockIdx.x * 4 + wave) * 64;   // 64 output positions per wave
    const int m     = ctile * 16 + lm;                // output channel (A row)

    const __bf16* inb  = in + (size_t)b * Cin * Lin;
    const __bf16* wrow = wbf + (size_t)m * KT;        // K = ci*3+t contiguous

    __builtin_prefetch(&inb[lbase * stride], 0, 3);   // global_prefetch_b8

    v8f acc0 = {}, acc1 = {}, acc2 = {}, acc3 = {};
    for (int k0 = 0; k0 < KT; k0 += 32) {
        // A fragment: each VGPR holds two consecutive K -> one packed b32 load
        union { unsigned u[8]; v16bf v; } A;
        #pragma unroll
        for (int j = 0; j < 8; ++j) {
            const int kk = k0 + 2 * j + 8 * h + ((j >= 4) ? 8 : 0);
            A.u[j] = (kk < KT) ? *(const unsigned*)(wrow + kk) : 0u;  // KT even, kk even
        }
        #pragma unroll
        for (int s = 0; s < 4; ++s) {
            const int n = lbase + s * 16 + lm;        // output position (B column)
            v16bf bfrag;
            #pragma unroll
            for (int e = 0; e < 16; ++e) {
                const int kk = k0 + kmapB(e, h);
                __bf16 bv = (__bf16)0.f;
                if (kk < KT) {
                    const int ci = kk / 3, t = kk - 3 * ci;
                    const int li = n * stride + t - 1;
                    if (li >= 0 && li < Lin) bv = inb[(size_t)ci * Lin + li];
                }
                bfrag[e] = bv;
            }
            v8f& acc = (s == 0) ? acc0 : (s == 1) ? acc1 : (s == 2) ? acc2 : acc3;
            acc = __builtin_amdgcn_wmma_f32_16x16x32_bf16(
                false, A.v, false, bfrag, (short)0, acc, false, false);
        }
    }
    // Epilogue: BN + ReLU, D layout: row = r + 8h, col = lm
    #pragma unroll
    for (int s = 0; s < 4; ++s) {
        const v8f& acc = (s == 0) ? acc0 : (s == 1) ? acc1 : (s == 2) ? acc2 : acc3;
        const int n = lbase + s * 16 + lm;
        #pragma unroll
        for (int r = 0; r < 8; ++r) {
            const int co = ctile * 16 + r + h * 8;
            const float sc = g[co] * __frsqrt_rn(vv[co] + EPS);
            const float val = (acc[r] - mm[co]) * sc + bb[co];
            out[((size_t)b * Cout + co) * Lout + n] = (__bf16)fmaxf(val, 0.f);
        }
    }
}

// ---------------- 4) mean pool over L (one wave per (b,c) row), bf16 in ----------
__global__ __launch_bounds__(256)
void meanpool(const __bf16* __restrict__ in, float* __restrict__ out, int BC, int L)
{
    const int wid  = (blockIdx.x * 256 + threadIdx.x) >> 5;
    const int lane = threadIdx.x & 31;
    if (wid >= BC) return;
    const __bf16* r = in + (size_t)wid * L;
    float s = 0.f;
    for (int i = lane; i < L; i += 32) s += (float)r[i];
    #pragma unroll
    for (int o = 16; o > 0; o >>= 1) s += __shfl_xor(s, o, 32);
    if (lane == 0) out[wid] = s / (float)L;
}

// ---------------- 5) FC: out[m,n] = sum_k A[m,k]*W[n,k] + bias[n] (bf16 WMMA) ------
__global__ __launch_bounds__(32)
void gemm_wmma_bias(const float* __restrict__ A, const float* __restrict__ W,
                    const float* __restrict__ bias, float* __restrict__ out,
                    int N, int K)
{
    const int mtile = blockIdx.x, ntile = blockIdx.y;
    const int lane = threadIdx.x, lm = lane & 15, h = lane >> 4;
    v8f acc = {};
    for (int k0 = 0; k0 < K; k0 += 32) {
        v16bf afrag, bfrag;
        #pragma unroll
        for (int e = 0; e < 16; ++e) {
            const int ka = k0 + kmapA(e, h);
            const int kb = k0 + kmapB(e, h);
            afrag[e] = f2bf((ka < K) ? A[(size_t)(mtile * 16 + lm) * K + ka] : 0.f);
            bfrag[e] = f2bf((kb < K) ? W[(size_t)(ntile * 16 + lm) * K + kb] : 0.f);
        }
        acc = __builtin_amdgcn_wmma_f32_16x16x32_bf16(
            false, afrag, false, bfrag, (short)0, acc, false, false);
    }
    const int n = ntile * 16 + lm;
    #pragma unroll
    for (int r = 0; r < 8; ++r) {
        const int m = mtile * 16 + r + h * 8;
        out[(size_t)m * N + n] = acc[r] + bias[n];
    }
}

// ---------------- 6) decoder ConvTranspose1d (k=4, s=2, p=1) + BN + ReLU ----------
__global__ __launch_bounds__(256)
void dec_convt_bnrelu(const float* __restrict__ in, const float* __restrict__ w,
                      const float* __restrict__ g, const float* __restrict__ bb,
                      const float* __restrict__ mm, const float* __restrict__ vv,
                      float* __restrict__ out, int B, int Cin, int Cout, int Lin, int Lout)
{
    const int idx = blockIdx.x * 256 + threadIdx.x;
    const int total = B * Cout * Lout;
    if (idx >= total) return;
    const int lo = idx % Lout;
    const int co = (idx / Lout) % Cout;
    const int b  = idx / (Lout * Cout);
    const float* xr = in + (size_t)b * Cin * Lin;
    float acc = 0.f;
    #pragma unroll
    for (int t = 0; t < 4; ++t) {
        const int num = lo + 1 - t;
        if (num & 1) continue;
        const int li = num >> 1;
        if (li < 0 || li >= Lin) continue;
        for (int ci = 0; ci < Cin; ++ci)
            acc += w[((size_t)ci * Cout + co) * 4 + t] * xr[(size_t)ci * Lin + li];
    }
    const float s = g[co] * __frsqrt_rn(vv[co] + EPS);
    out[idx] = fmaxf((acc - mm[co]) * s + bb[co], 0.f);
}

// ---------------- 7) decoder conv (k=3, pad=1): BN+ReLU or bias-only --------------
__global__ __launch_bounds__(256)
void dec_conv3(const float* __restrict__ in, const float* __restrict__ w,
               const float* __restrict__ g, const float* __restrict__ bb,
               const float* __restrict__ mm, const float* __restrict__ vv,
               const float* __restrict__ bias, float* __restrict__ out,
               int B, int Cin, int Cout, int L, int bnrelu)
{
    const int idx = blockIdx.x * 256 + threadIdx.x;
    const int total = B * Cout * L;
    if (idx >= total) return;
    const int l  = idx % L;
    const int co = (idx / L) % Cout;
    const int b  = idx / (L * Cout);
    const float* xr = in + (size_t)b * Cin * L;
    float acc = 0.f;
    for (int ci = 0; ci < Cin; ++ci) {
        const float* wr = w + ((size_t)co * Cin + ci) * 3;
        const float* xx = xr + (size_t)ci * L;
        if (l > 0)     acc += wr[0] * xx[l - 1];
                       acc += wr[1] * xx[l];
        if (l < L - 1) acc += wr[2] * xx[l + 1];
    }
    float val;
    if (bnrelu) {
        const float s = g[co] * __frsqrt_rn(vv[co] + EPS);
        val = fmaxf((acc - mm[co]) * s + bb[co], 0.f);
    } else {
        val = acc + bias[co];
    }
    out[idx] = val;
}

// =======================================================================
extern "C" void kernel_launch(void* const* d_in, const int* in_sizes, int n_in,
                              void* d_out, int out_size, void* d_ws, size_t ws_size,
                              hipStream_t stream)
{
    (void)in_sizes; (void)n_in; (void)out_size; (void)ws_size;

    const int B = 256, L0 = 8192, NP = 512, KEEP = 128;

    const float* x     = (const float*)d_in[0];
    const float* noise = (const float*)d_in[1];

    // encoder params (setup_inputs() insertion order; BN dicts are g,b,m,v)
    const float* conv1_w = (const float*)d_in[2];
    const float* bn0[4]  = {(const float*)d_in[3], (const float*)d_in[4],
                            (const float*)d_in[5], (const float*)d_in[6]};
    auto EW = [&](int i, int which) -> const float* {         // which: 0=w1, 1=w2
        return (const float*)d_in[7 + 10 * (i - 1) + (which ? 5 : 0)];
    };
    auto EBN = [&](int i, int which, int j) -> const float* { // j: 0=g,1=b,2=m,3=v
        return (const float*)d_in[7 + 10 * (i - 1) + (which ? 6 : 1) + j];
    };
    const float* enc_fc_w = (const float*)d_in[47];
    const float* enc_fc_b = (const float*)d_in[48];
    const float* dec_fc_w = (const float*)d_in[49];
    const float* dec_fc_b = (const float*)d_in[50];
    auto DW = [&](int i, int which) -> const float* {
        return (const float*)d_in[51 + 10 * (i - 1) + (which ? 5 : 0)];
    };
    auto DBN = [&](int i, int which, int j) -> const float* {
        return (const float*)d_in[51 + 10 * (i - 1) + (which ? 6 : 1) + j];
    };
    const float* final_w = (const float*)d_in[91];
    const float* final_b = (const float*)d_in[92];

    // d_out slices: recon(4096) | masked_x(2097152) | mask(2097152) | latent(32768)
    float* out_f  = (float*)d_out;
    float* recon  = out_f;
    float* masked = out_f + 4096;
    float* maskp  = masked + (size_t)B * L0;
    float* latent = maskp  + (size_t)B * L0;

    // workspace (bf16 encoder activations): ~136 MB
    char* wsb = (char*)d_ws;
    __bf16* bufA = (__bf16*)wsb;                                   // 33,554,432 bf16
    __bf16* bufB = (__bf16*)(wsb + (size_t)67108864);              // 16,777,216 bf16
    __bf16* bufC = (__bf16*)(wsb + (size_t)67108864 + 33554432);   // 16,777,216 bf16
    __bf16* WBF  = (__bf16*)(wsb + (size_t)134217728);             // 98,304 bf16 weights
    float*  G    = (float*) (wsb + (size_t)134217728 + 262144);    // 32768 f32
    float*  D0   = G + 32768;
    float*  D1a  = D0  + 32768;  float* D1b = D1a + 32768;
    float*  D2a  = D1b + 32768;  float* D2b = D2a + 32768;
    float*  D3a  = D2b + 32768;  float* D3b = D3a + 32768;
    float*  D4a  = D3b + 32768;  float* D4b = D4a + 32768;

    // ---- 0) convert the 8 encoder conv weights to bf16 (contiguous K=ci*3+t rows) --
    const int wsz[8]  = {768, 768, 1536, 3072, 6144, 12288, 24576, 49152};
    const int woff[8] = {0, 768, 1536, 3072, 6144, 12288, 24576, 49152};
    {
        int li = 0;
        for (int i = 1; i <= 4; ++i)
            for (int which = 0; which < 2; ++which, ++li)
                f32_to_bf16<<<dim3((wsz[li] + 255) / 256), dim3(256), 0, stream>>>(
                    EW(i, which), WBF + woff[li], wsz[li]);
    }

    // ---- 1) mask + masked_x (written straight into d_out) ----
    mask_kernel<<<dim3(B), dim3(256), 0, stream>>>(x, noise, masked, maskp, L0, NP, KEEP);

    // ---- 2) conv1 (1->16, k=7) + BN + ReLU, bf16 out ----
    conv1_bnrelu<<<dim3(L0 / 256, B), dim3(256), 0, stream>>>(
        masked, conv1_w, bn0[0], bn0[1], bn0[2], bn0[3], bufA, L0);

    // ---- 3) encoder WMMA conv stack (bf16 in/out) ----
    struct LayerDesc { int li, i, which, Cin, Cout, Lin, Lout, stride;
                       const __bf16 *in; __bf16 *out; };
    const LayerDesc layers[8] = {
        {0, 1, 0,  16,  16, 8192, 4096, 2, bufA, bufB},
        {1, 1, 1,  16,  16, 4096, 4096, 1, bufB, bufC},
        {2, 2, 0,  16,  32, 4096, 2048, 2, bufC, bufB},
        {3, 2, 1,  32,  32, 2048, 2048, 1, bufB, bufC},
        {4, 3, 0,  32,  64, 2048, 1024, 2, bufC, bufB},
        {5, 3, 1,  64,  64, 1024, 1024, 1, bufB, bufC},
        {6, 4, 0,  64, 128, 1024,  512, 2, bufC, bufB},
        {7, 4, 1, 128, 128,  512,  512, 1, bufB, bufC},
    };
    for (int s = 0; s < 8; ++s) {
        const LayerDesc& Ld = layers[s];
        conv_wmma_bnrelu<<<dim3(Ld.Lout / 256, Ld.Cout / 16, B), dim3(128), 0, stream>>>(
            Ld.in, WBF + woff[Ld.li],
            EBN(Ld.i, Ld.which, 0), EBN(Ld.i, Ld.which, 1),
            EBN(Ld.i, Ld.which, 2), EBN(Ld.i, Ld.which, 3),
            Ld.out, Ld.Cin, Ld.Cout, Ld.Lin, Ld.Lout, Ld.stride);
    }

    // ---- 4) mean pool (256*128 rows of 512, bf16 in) ----
    meanpool<<<dim3((B * 128) / 8), dim3(256), 0, stream>>>(bufC, G, B * 128, 512);

    // ---- 5) encoder FC -> latent (in d_out); decoder FC -> D0 ----
    gemm_wmma_bias<<<dim3(B / 16, 128 / 16), dim3(32), 0, stream>>>(G,      enc_fc_w, enc_fc_b, latent, 128, 128);
    gemm_wmma_bias<<<dim3(B / 16, 128 / 16), dim3(32), 0, stream>>>(latent, dec_fc_w, dec_fc_b, D0,     128, 128);

    // ---- 6) decoder (tiny; scalar f32) ----
    struct UpDesc { int i, Cin, Cout, Lin; const float *in; float *mid, *out; };
    const UpDesc ups[4] = {
        {1, 128,  64, 1, D0,  D1a, D1b},
        {2,  64,  32, 2, D1b, D2a, D2b},
        {3,  32,  16, 4, D2b, D3a, D3b},
        {4,  16,   8, 8, D3b, D4a, D4b},
    };
    for (int s = 0; s < 4; ++s) {
        const UpDesc& U = ups[s];
        const int Lout = U.Lin * 2;
        const int tot = B * U.Cout * Lout;
        dec_convt_bnrelu<<<dim3((tot + 255) / 256), dim3(256), 0, stream>>>(
            U.in, DW(U.i, 0), DBN(U.i, 0, 0), DBN(U.i, 0, 1), DBN(U.i, 0, 2), DBN(U.i, 0, 3),
            U.mid, B, U.Cin, U.Cout, U.Lin, Lout);
        dec_conv3<<<dim3((tot + 255) / 256), dim3(256), 0, stream>>>(
            U.mid, DW(U.i, 1), DBN(U.i, 1, 0), DBN(U.i, 1, 1), DBN(U.i, 1, 2), DBN(U.i, 1, 3),
            final_b, U.out, B, U.Cout, U.Cout, Lout, 1);
    }

    // ---- 7) final conv (8->1, k=3) + bias -> recon in d_out ----
    {
        const int tot = B * 1 * 16;
        dec_conv3<<<dim3((tot + 255) / 256), dim3(256), 0, stream>>>(
            D4b, final_w, final_b, final_b, final_b, final_b, final_b,
            recon, B, 8, 1, 16, 0);
    }
}
